// BFAN_53609781789105
// MI455X (gfx1250) — compile-verified, matrix operand order checked by reading, not run on hardware
//
#include <hip/hip_runtime.h>
#include <hip/hip_bf16.h>

// ---------------------------------------------------------------------------
// BFAN similarity (t2i + i2t focal attention) for MI455X / gfx1250.
// Core GEMM (raw[r][w] = img_r . cap_w) runs on v_wmma_f32_16x16x32_bf16.
// wctx GEMMs are eliminated via Gram-matrix quadratic forms:
//   num      = sum_r re[w,r]*raw[r,w]
//   |wctx|^2 = re^T G re,  G = X X^T  (precomputed, L2-resident)
// Image bf16 copies are PADDED to 48 rows (zeros) so all WMMA fragment loads
// are branch-free / unconditional.
// ---------------------------------------------------------------------------

typedef __attribute__((ext_vector_type(16))) __bf16 v16bf;
typedef __attribute__((ext_vector_type(8)))  __bf16 v8bf;
typedef __attribute__((ext_vector_type(8)))  float  v8f;

#define N_IMAGE   128
#define N_REGIONS 36
#define R_PAD     48          // 36 padded to 3 WMMA tiles
#define N_CAPTION 128
#define N_WORD    32
#define DDIM      1024

// Load one 16x32 bf16 WMMA fragment (A layout; B is K x N column-major, so a
// B fragment loads rows of the source matrix with the identical pattern).
// ISA 16-bit A layout: lanes 0-15 hold M=lane,    K = {k0..+7,  k0+16..+23};
//                      lanes 16-31 hold M=lane-16, K = {k0+8..+15, k0+24..+31}.
static __device__ __forceinline__ v16bf load_frag(const __bf16* __restrict__ rowPtr, int k0) {
    v8bf lo = *reinterpret_cast<const v8bf*>(rowPtr + k0);
    v8bf hi = *reinterpret_cast<const v8bf*>(rowPtr + k0 + 16);
    return __builtin_shufflevector(lo, hi, 0,1,2,3,4,5,6,7,8,9,10,11,12,13,14,15);
}

// Per-lane row base pointer for a fragment whose 16 rows start at rowBase.
static __device__ __forceinline__ const __bf16* frag_base(const __hip_bfloat16* X,
                                                          int rowBase, int lane) {
    const __bf16* p = reinterpret_cast<const __bf16*>(X);
    return p + (size_t)(rowBase + (lane & 15)) * DDIM + ((lane >> 4) << 3);
}

// ---------------------------------------------------------------------------
// Kernel 1: f32 -> bf16 conversion + per-row L2 norms. One block per row.
// Image rows are written into the 48-row padded layout; padding rows zeroed.
// Grid: 128*48 image rows (incl. padding) + 128*32 caption rows.
// ---------------------------------------------------------------------------
__global__ void __launch_bounds__(128) bfan_convert_norms(
    const float* __restrict__ img, const float* __restrict__ cap,
    __hip_bfloat16* __restrict__ imgP, __hip_bfloat16* __restrict__ capB,
    float* __restrict__ imgN, float* __restrict__ capN)
{
    const int row = blockIdx.x;
    const int tid = threadIdx.x;

    const float* src;
    __hip_bfloat16* dst;
    float* nrm;
    if (row < N_IMAGE * R_PAD) {
        const int i = row / R_PAD, r = row % R_PAD;
        dst = imgP + (size_t)row * DDIM;
        if (r >= N_REGIONS) {                 // zero padding row
            const __hip_bfloat16 z = __float2bfloat16(0.f);
            for (int j = tid; j < DDIM; j += 128) dst[j] = z;
            return;
        }
        src = img + ((size_t)i * N_REGIONS + r) * DDIM;
        nrm = imgN + i * N_REGIONS + r;
    } else {
        const int r = row - N_IMAGE * R_PAD;
        src = cap + (size_t)r * DDIM;
        dst = capB + (size_t)r * DDIM;
        nrm = capN + r;
    }

    float s = 0.f;
    for (int j = tid; j < DDIM; j += 128) {
        float x = src[j];
        s += x * x;
        dst[j] = __float2bfloat16(x);
    }
    for (int o = 16; o > 0; o >>= 1) s += __shfl_down(s, o, 32);
    __shared__ float ps[4];
    if ((tid & 31) == 0) ps[tid >> 5] = s;
    __syncthreads();
    if (tid == 0) *nrm = sqrtf(ps[0] + ps[1] + ps[2] + ps[3]);
}

// ---------------------------------------------------------------------------
// Kernel 2: Gram matrices G = X X^T via WMMA bf16. One block per image/caption.
// rowsPad is a multiple of 16 (48 for images, 32 for captions), so all loads
// are unconditional; only the final store is masked to rows x rows.
// ---------------------------------------------------------------------------
__global__ void __launch_bounds__(128) bfan_gram(
    const __hip_bfloat16* __restrict__ Xall, float* __restrict__ Gall,
    int rows, int rowsPad, int tpd)
{
    const int b    = blockIdx.x;
    const int tid  = threadIdx.x;
    const int lane = tid & 31;
    const int wave = tid >> 5;
    const __hip_bfloat16* X = Xall + (size_t)b * rowsPad * DDIM;
    float* G = Gall + (size_t)b * rows * rows;
    const int ntiles = tpd * tpd;
    for (int t = wave; t < ntiles; t += 4) {
        const int tm = t / tpd, tn = t % tpd;
        const __bf16* pa = frag_base(X, tm * 16, lane);
        const __bf16* pb = frag_base(X, tn * 16, lane);
        v8f acc = {};
        #pragma unroll 4
        for (int k0 = 0; k0 < DDIM; k0 += 32) {
            v16bf a  = load_frag(pa, k0);
            v16bf bb = load_frag(pb, k0);
            acc = __builtin_amdgcn_wmma_f32_16x16x32_bf16(false, a, false, bb,
                                                          (short)0, acc, false, false);
        }
        const int n  = tn * 16 + (lane & 15);
        const int mb = tm * 16 + ((lane >> 4) << 3);
        #pragma unroll
        for (int v = 0; v < 8; ++v) {
            const int m = mb + v;
            if (m < rows && n < rows) G[m * rows + n] = acc[v];
        }
    }
}

// ---------------------------------------------------------------------------
// Kernel 3: main BFAN kernel. One block per (caption c, image i) pair.
// 192 threads = 6 waves. Phase 1: raw (48x32 padded) via WMMA, one tile/wave,
// fully branch-free inner loop. Phase 2: focal attention both directions +
// cosine via Gram quadratic forms.
// ---------------------------------------------------------------------------
__global__ void __launch_bounds__(192) bfan_main(
    const __hip_bfloat16* __restrict__ imgP, const __hip_bfloat16* __restrict__ capB,
    const float* __restrict__ GIall, const float* __restrict__ GCall,
    const float* __restrict__ imgN, const float* __restrict__ capN,
    float* __restrict__ out)
{
    const int c = blockIdx.x, i = blockIdx.y;
    const int tid = threadIdx.x, lane = tid & 31, wave = tid >> 5;

    __shared__ float raw[N_REGIONS][33];   // raw[r][w], padded (odd stride)
    __shared__ float aL [N_REGIONS][33];   // leaky_relu(raw)
    __shared__ float rn1[N_REGIONS];       // ||aL[r,:]|| + 1e-8   (t2i l2norm)
    __shared__ float rn2[N_WORD];          // ||aL[:,w]|| + 1e-8   (i2t l2norm)
    __shared__ float reT[N_WORD][37];      // t2i attention, per word w
    __shared__ float reI[N_REGIONS][33];   // i2t attention, per region r
    __shared__ float sOut[3];

    const float* GI = GIall + (size_t)i * N_REGIONS * N_REGIONS;
    const float* GC = GCall + (size_t)c * N_WORD * N_WORD;
    // Warm L2 for the epilogue while the WMMA phase runs.
    __builtin_prefetch(GI, 0, 1);
    __builtin_prefetch(GC, 0, 1);

    // ---- Phase 1: raw = img_i (48x1024 padded) @ cap_c^T (1024x32) ----
    {
        const __hip_bfloat16* Xi = imgP + (size_t)i * R_PAD * DDIM;
        const __hip_bfloat16* Xc = capB + (size_t)c * N_WORD * DDIM;
        const int tm = wave >> 1, tn = wave & 1;       // 3x2 tile grid
        const __bf16* pa = frag_base(Xi, tm * 16, lane);
        const __bf16* pb = frag_base(Xc, tn * 16, lane);
        v8f acc = {};
        #pragma unroll 4
        for (int k0 = 0; k0 < DDIM; k0 += 32) {
            v16bf a = load_frag(pa, k0);
            v16bf b = load_frag(pb, k0);
            acc = __builtin_amdgcn_wmma_f32_16x16x32_bf16(false, a, false, b,
                                                          (short)0, acc, false, false);
        }
        const int n  = tn * 16 + (lane & 15);
        const int mb = tm * 16 + ((lane >> 4) << 3);
        #pragma unroll
        for (int v = 0; v < 8; ++v) {
            const int m = mb + v;
            if (m < N_REGIONS) raw[m][n] = acc[v];
        }
    }
    __syncthreads();

    // ---- leaky relu + norms over w (per region) ----
    if (tid < N_REGIONS) {
        float s = 0.f;
        for (int w = 0; w < N_WORD; ++w) {
            float x = raw[tid][w];
            float a = x > 0.f ? x : 0.1f * x;
            aL[tid][w] = a;
            s += a * a;
        }
        rn1[tid] = sqrtf(s) + 1e-8f;
    }
    __syncthreads();
    // ---- norms over r (per word) ----
    if (tid < N_WORD) {
        float s = 0.f;
        for (int r = 0; r < N_REGIONS; ++r) { float a = aL[r][tid]; s += a * a; }
        rn2[tid] = sqrtf(s) + 1e-8f;
    }
    __syncthreads();

    if (wave == 0) {
        // ---- t2i: lane = word w; softmax/focal over regions (S = 36) ----
        const int w = lane;
        float mx = -1e30f;
        for (int r = 0; r < N_REGIONS; ++r) {
            float v = 20.f * aL[r][w] / rn1[r];
            reT[w][r] = v;
            mx = fmaxf(mx, v);
        }
        float s = 0.f;
        for (int r = 0; r < N_REGIONS; ++r) { float e = __expf(reT[w][r] - mx); reT[w][r] = e; s += e; }
        float inv = 1.f / s, s2 = 0.f;
        for (int r = 0; r < N_REGIONS; ++r) {
            float a = reT[w][r] * inv;                       // softmax attn
            a = (a * 36.f - 1.f > 0.f) ? a : 0.f;           // focal_equal mask
            reT[w][r] = a; s2 += a;
        }
        float inv2 = 1.f / s2, num = 0.f;
        for (int r = 0; r < N_REGIONS; ++r) {
            float re = reT[w][r] * inv2;
            reT[w][r] = re;
            num += re * raw[r][w];                          // cap_w . wctx_w
        }
        float q = 0.f;
        for (int r = 0; r < N_REGIONS; ++r) {
            const float* Gr = GI + r * N_REGIONS;
            float t = 0.f;
            for (int r2 = 0; r2 < N_REGIONS; ++r2) t += Gr[r2] * reT[w][r2];
            q += reT[w][r] * t;                             // ||wctx_w||^2
        }
        float wn = fmaxf(sqrtf(fmaxf(q, 0.f)), 1e-8f);
        float cn = fmaxf(capN[c * N_WORD + w], 1e-8f);
        float cw = num / (cn * wn);
        for (int o = 16; o > 0; o >>= 1) cw += __shfl_down(cw, o, 32);
        if (lane == 0) sOut[0] = cw * (1.f / 32.f);
    } else if (wave <= 2) {
        // ---- i2t: lane maps to region r; softmax/focal over words (S = 32) ----
        const int r = tid - 32;                             // 0..63, valid < 36
        float cr = 0.f;
        if (r < N_REGIONS) {
            float mx = -1e30f;
            for (int w = 0; w < N_WORD; ++w) {
                float v = 20.f * aL[r][w] / rn2[w];
                reI[r][w] = v;
                mx = fmaxf(mx, v);
            }
            float s = 0.f;
            for (int w = 0; w < N_WORD; ++w) { float e = __expf(reI[r][w] - mx); reI[r][w] = e; s += e; }
            float inv = 1.f / s, s2 = 0.f;
            for (int w = 0; w < N_WORD; ++w) {
                float a = reI[r][w] * inv;
                a = (a * 32.f - 1.f > 0.f) ? a : 0.f;
                reI[r][w] = a; s2 += a;
            }
            float inv2 = 1.f / s2, num = 0.f;
            for (int w = 0; w < N_WORD; ++w) {
                float re = reI[r][w] * inv2;
                reI[r][w] = re;
                num += re * raw[r][w];                      // img_r . wctx_r
            }
            float q = 0.f;
            for (int w = 0; w < N_WORD; ++w) {
                const float* Gw = GC + w * N_WORD;
                float t = 0.f;
                for (int w2 = 0; w2 < N_WORD; ++w2) t += Gw[w2] * reI[r][w2];
                q += reI[r][w] * t;                         // ||wctx_r||^2
            }
            float wn = fmaxf(sqrtf(fmaxf(q, 0.f)), 1e-8f);
            float in = fmaxf(imgN[i * N_REGIONS + r], 1e-8f);
            cr = num / (in * wn);
        }
        for (int o = 16; o > 0; o >>= 1) cr += __shfl_down(cr, o, 32);
        if (lane == 0) sOut[wave] = cr;
    }
    __syncthreads();
    if (tid == 0)
        out[(size_t)i * N_CAPTION + c] = sOut[0] + (sOut[1] + sOut[2]) * (1.f / 36.f);
}

// ---------------------------------------------------------------------------
// Launch
// ---------------------------------------------------------------------------
extern "C" void kernel_launch(void* const* d_in, const int* in_sizes, int n_in,
                              void* d_out, int out_size, void* d_ws, size_t ws_size,
                              hipStream_t stream) {
    const float* images   = (const float*)d_in[0];   // (128, 36, 1024) f32
    const float* captions = (const float*)d_in[1];   // (128, 32, 1024) f32
    // d_in[2] = cap_lens, all == 32: unused (vectorized exactly like reference)

    // Workspace layout (all 16B aligned):
    constexpr size_t IMG_E = (size_t)N_IMAGE * R_PAD * DDIM;         // bf16, padded
    constexpr size_t CAP_E = (size_t)N_CAPTION * N_WORD * DDIM;      // bf16
    char* ws = (char*)d_ws;
    __hip_bfloat16* imgP = (__hip_bfloat16*)ws;
    __hip_bfloat16* capB = (__hip_bfloat16*)(ws + IMG_E * 2);
    float* G_I  = (float*)(ws + IMG_E * 2 + CAP_E * 2);
    float* G_C  = G_I + (size_t)N_IMAGE * N_REGIONS * N_REGIONS;
    float* imgN = G_C + (size_t)N_CAPTION * N_WORD * N_WORD;
    float* capN = imgN + (size_t)N_IMAGE * N_REGIONS;

    // 1) bf16 copies (48-row padded images) + row norms
    bfan_convert_norms<<<N_IMAGE * R_PAD + N_CAPTION * N_WORD, 128, 0, stream>>>(
        images, captions, imgP, capB, imgN, capN);

    // 2) Gram matrices (WMMA bf16)
    bfan_gram<<<N_IMAGE,   128, 0, stream>>>(imgP, G_I, N_REGIONS, R_PAD, 3);
    bfan_gram<<<N_CAPTION, 128, 0, stream>>>(capB, G_C, N_WORD, N_WORD,   2);

    // 3) main fused kernel over all (c, i) pairs
    dim3 grid(N_CAPTION, N_IMAGE);
    bfan_main<<<grid, 192, 0, stream>>>(imgP, capB, G_I, G_C, imgN, capN, (float*)d_out);
}
